// TransformerEncoderLayer_67319317397624
// MI455X (gfx1250) — matrix operand order repaired and checked by reading, not compiled, not run on hardware
//
#include <hip/hip_runtime.h>
#include <hip/hip_bf16.h>

// ---------------------------------------------------------------------------
// MI455X transformer encoder layer, bf16 WMMA everywhere.
// B=2 N=2048 C=768 H=12 D=64 HID=3072. M = B*N = 4096 rows for all GEMMs.
// All operands are L2-resident (192MB); wave tiles sized for L2-bandwidth
// arithmetic intensity: 64x64 (32 FLOP/B) wide GEMMs, 64x32 narrow ones.
// Attention uses 64-key chunks to amortize softmax shuffle/hazard overhead.
// ---------------------------------------------------------------------------

typedef __bf16 bf16;
typedef __attribute__((ext_vector_type(16))) __bf16 v16bf;
typedef __attribute__((ext_vector_type(8)))  __bf16 bf16x8;
typedef __attribute__((ext_vector_type(8)))  float  v8f;

__device__ __forceinline__ v8f wmma_bf16(v16bf a, v16bf b, v8f c) {
  return __builtin_amdgcn_wmma_f32_16x16x32_bf16(false, a, false, b, (short)0, c,
                                                 false, false);
}

// A-fragment (16x32, MxK): lane holds row m=lane&15; element e holds
// k = (e<8 ? e : e+8) + 8*(lane>=16).  Two contiguous 16B loads per lane.
__device__ __forceinline__ v16bf load_a(const bf16* p) {
  bf16x8 lo = *(const bf16x8*)p;
  bf16x8 hi = *(const bf16x8*)(p + 16);
  v16bf r;
#pragma unroll
  for (int i = 0; i < 8; ++i) { r[i] = lo[i]; r[8 + i] = hi[i]; }
  return r;
}

// B-fragment (32x16, KxN): lane holds col n=lane&15; element e holds
// k = (e<8 ? e : e+8) + 8*(lane>=16).  Two contiguous 16B loads per lane.
__device__ __forceinline__ v16bf load_b(const bf16* p) {
  return load_a(p);
}

// ---------------------------------------------------------------------------
// Weight convert + transpose: in f32 [K,N] -> out bf16 [N,K]
// ---------------------------------------------------------------------------
__global__ void __launch_bounds__(256)
conv_t_kernel(const float* __restrict__ in, bf16* __restrict__ out, int K, int N) {
  int i = blockIdx.x * 256 + threadIdx.x;
  if (i >= K * N) return;
  int k = i / N;
  int n = i - k * N;
  out[(size_t)n * K + k] = (bf16)in[i];
}

// ---------------------------------------------------------------------------
// LayerNorm over C=768, one row per block (256 threads, 3 elems/thread),
// output bf16.
// ---------------------------------------------------------------------------
__global__ void __launch_bounds__(256)
ln_kernel(const float* __restrict__ x, const float* __restrict__ g,
          const float* __restrict__ b, bf16* __restrict__ out) {
  const int C = 768;
  const int t = threadIdx.x;
  const float* xr = x + (size_t)blockIdx.x * C;
  float v0 = xr[t], v1 = xr[t + 256], v2 = xr[t + 512];

  __shared__ float sm[8];
  float s = v0 + v1 + v2;
#pragma unroll
  for (int o = 16; o > 0; o >>= 1) s += __shfl_xor(s, o);
  if ((t & 31) == 0) sm[t >> 5] = s;
  __syncthreads();
  float tot = sm[0] + sm[1] + sm[2] + sm[3] + sm[4] + sm[5] + sm[6] + sm[7];
  float mu = tot * (1.0f / 768.0f);
  float d0 = v0 - mu, d1 = v1 - mu, d2 = v2 - mu;
  float s2 = d0 * d0 + d1 * d1 + d2 * d2;
  __syncthreads();
#pragma unroll
  for (int o = 16; o > 0; o >>= 1) s2 += __shfl_xor(s2, o);
  if ((t & 31) == 0) sm[t >> 5] = s2;
  __syncthreads();
  float var = (sm[0] + sm[1] + sm[2] + sm[3] + sm[4] + sm[5] + sm[6] + sm[7]) *
              (1.0f / 768.0f);
  float rstd = rsqrtf(var + 1e-5f);
  bf16* orow = out + (size_t)blockIdx.x * C;
  orow[t]       = (bf16)(d0 * rstd * g[t]       + b[t]);
  orow[t + 256] = (bf16)(d1 * rstd * g[t + 256] + b[t + 256]);
  orow[t + 512] = (bf16)(d2 * rstd * g[t + 512] + b[t + 512]);
}

// ---------------------------------------------------------------------------
// Generic bf16 GEMM:  C[M,N] = A[M,K] @ Bt[N,K]^T  (+ compile-time epilogue)
// Block = 128 threads = 4 waves (2x2); wave tile = 16*TI x 16*TJ.
// EPI 0: QKV scatter into q (scaled 1/8), k, v^T buffers
// EPI 1: out_f32 = acc + bias + resid          (f32 residual add)
// EPI 2: out_bf  = bf16(gelu_exact(acc+bias))  (MLP up-proj)
// ---------------------------------------------------------------------------
template <int TI, int TJ, int EPI>
__global__ void __launch_bounds__(128)
gemm_wmma(const bf16* __restrict__ A, const bf16* __restrict__ Bt,
          const float* __restrict__ bias, int M, int N, int K,
          bf16* __restrict__ out_bf, float* __restrict__ out_f,
          const float* __restrict__ resid,
          bf16* __restrict__ qout, bf16* __restrict__ kout,
          bf16* __restrict__ vtout) {
  const int lane = threadIdx.x & 31;
  const int wave = threadIdx.x >> 5;
  const int hi   = lane >> 4;      // half-wave select
  const int ln16 = lane & 15;
  const int wr = wave >> 1, wc = wave & 1;
  const int rowBase = blockIdx.y * (32 * TI) + wr * (16 * TI);
  const int colBase = blockIdx.x * (32 * TJ) + wc * (16 * TJ);

  v8f acc[TI][TJ];
#pragma unroll
  for (int i = 0; i < TI; ++i)
#pragma unroll
    for (int j = 0; j < TJ; ++j) acc[i][j] = (v8f){};

  const bf16* arow[TI];
  const bf16* brow[TJ];
#pragma unroll
  for (int i = 0; i < TI; ++i)
    arow[i] = A + (size_t)(rowBase + 16 * i + ln16) * K + 8 * hi;
#pragma unroll
  for (int j = 0; j < TJ; ++j)
    brow[j] = Bt + (size_t)(colBase + 16 * j + ln16) * K + 8 * hi;

  for (int k0 = 0; k0 < K; k0 += 32) {
    v16bf af[TI], bfr[TJ];
#pragma unroll
    for (int i = 0; i < TI; ++i) af[i] = load_a(arow[i] + k0);
#pragma unroll
    for (int j = 0; j < TJ; ++j) bfr[j] = load_b(brow[j] + k0);
    __builtin_prefetch(arow[0] + k0 + 64, 0, 1);
    __builtin_prefetch(brow[0] + k0 + 64, 0, 1);
#pragma unroll
    for (int i = 0; i < TI; ++i)
#pragma unroll
      for (int j = 0; j < TJ; ++j)
        acc[i][j] = wmma_bf16(af[i], bfr[j], acc[i][j]);
  }

#pragma unroll
  for (int ti = 0; ti < TI; ++ti) {
#pragma unroll
    for (int tj = 0; tj < TJ; ++tj) {
#pragma unroll
      for (int r = 0; r < 8; ++r) {
        int row = rowBase + ti * 16 + r + 8 * hi;
        int col = colBase + tj * 16 + ln16;
        float v = acc[ti][tj][r] + bias[col];
        if (EPI == 0) {
          // col = h*192 + d*3 + s  (qkv reshape [N,H,D,3])
          int s = col % 3, hd = col / 3;
          int hh = hd >> 6, d = hd & 63;
          int bidx = row >> 11, n = row & 2047;
          size_t bh = (size_t)bidx * 12 + hh;
          if (s == 0)      qout[(bh * 2048 + n) * 64 + d] = (bf16)(v * 0.125f);
          else if (s == 1) kout[(bh * 2048 + n) * 64 + d] = (bf16)v;
          else             vtout[(bh * 64 + d) * 2048 + n] = (bf16)v;
        } else if (EPI == 1) {
          size_t idx = (size_t)row * N + col;
          out_f[idx] = v + resid[idx];
        } else {
          float ge = 0.5f * v * (1.0f + erff(v * 0.70710678118654752f));
          out_bf[(size_t)row * N + col] = (bf16)ge;
        }
      }
    }
  }
}

// ---------------------------------------------------------------------------
// Flash attention: grid(B*H, N/128), block 128 = 4 waves, each wave owns a
// 32-query tile (two 16-row WMMA tiles sharing K/V fragments) and walks keys
// in 64-wide chunks (halves softmax shuffle + WMMA->VALU hazard cost/key).
// q pre-scaled by 1/sqrt(D).  k row-major [n,d], v^T [d,n].
// P goes f32->bf16 through a per-wave LDS tile (32x64) to become A-fragments.
// ---------------------------------------------------------------------------
__global__ void __launch_bounds__(128)
attn_kernel(const bf16* __restrict__ q, const bf16* __restrict__ k,
            const bf16* __restrict__ vt, bf16* __restrict__ o, int N) {
  const int bh   = blockIdx.x;              // [0,24)
  const int br   = bh / 12, h = bh % 12;
  const int lane = threadIdx.x & 31;
  const int wave = threadIdx.x >> 5;
  const int hi   = lane >> 4;
  const int ln16 = lane & 15;
  const int q0   = blockIdx.y * 128 + wave * 32;

  const bf16* qb = q  + (size_t)bh * N * 64;
  const bf16* kb = k  + (size_t)bh * N * 64;
  const bf16* vb = vt + (size_t)bh * 64 * N;

  __shared__ __align__(16) bf16 plds[4][32 * 64];
  bf16* pw = plds[wave];

  v8f oacc[2][4];
  float mrow[2][8], lrow[2][8];
#pragma unroll
  for (int mt = 0; mt < 2; ++mt) {
#pragma unroll
    for (int t = 0; t < 4; ++t) oacc[mt][t] = (v8f){};
#pragma unroll
    for (int r = 0; r < 8; ++r) { mrow[mt][r] = -1e30f; lrow[mt][r] = 0.0f; }
  }

  // Persistent Q fragments: 2 m-tiles x 2 d-steps
  v16bf qa[2][2];
#pragma unroll
  for (int mt = 0; mt < 2; ++mt) {
    const bf16* qrow = qb + (size_t)(q0 + mt * 16 + ln16) * 64 + 8 * hi;
    qa[mt][0] = load_a(qrow);
    qa[mt][1] = load_a(qrow + 32);
  }

  for (int key0 = 0; key0 < N; key0 += 64) {
    // ---- S = (Q/sqrtD) K^T : 4 key tiles, K frags shared by both m-tiles.
    // Load K fragments in two groups of 2 tiles to bound register pressure.
    v8f s[2][4];
#pragma unroll
    for (int g = 0; g < 2; ++g) {
      const bf16* kr0 = kb + (size_t)(key0 + g * 32 + ln16) * 64 + 8 * hi;
      const bf16* kr1 = kb + (size_t)(key0 + g * 32 + 16 + ln16) * 64 + 8 * hi;
      v16bf kf00 = load_b(kr0), kf01 = load_b(kr0 + 32);
      v16bf kf10 = load_b(kr1), kf11 = load_b(kr1 + 32);
#pragma unroll
      for (int mt = 0; mt < 2; ++mt) {
        v8f t0 = wmma_bf16(qa[mt][0], kf00, (v8f){});
        s[mt][2 * g]     = wmma_bf16(qa[mt][1], kf01, t0);
        v8f t1 = wmma_bf16(qa[mt][0], kf10, (v8f){});
        s[mt][2 * g + 1] = wmma_bf16(qa[mt][1], kf11, t1);
      }
    }

    // ---- online softmax over this 64-key chunk, per m-tile ----------------
#pragma unroll
    for (int mt = 0; mt < 2; ++mt) {
#pragma unroll
      for (int r = 0; r < 8; ++r) {
        float a0 = s[mt][0][r], a1 = s[mt][1][r];
        float a2 = s[mt][2][r], a3 = s[mt][3][r];
        float cm = fmaxf(fmaxf(a0, a1), fmaxf(a2, a3));
        cm = fmaxf(cm, __shfl_xor(cm, 1));
        cm = fmaxf(cm, __shfl_xor(cm, 2));
        cm = fmaxf(cm, __shfl_xor(cm, 4));
        cm = fmaxf(cm, __shfl_xor(cm, 8));
        float mnew  = fmaxf(mrow[mt][r], cm);
        float scale = __expf(mrow[mt][r] - mnew);
        mrow[mt][r] = mnew;
        float p0 = __expf(a0 - mnew);
        float p1 = __expf(a1 - mnew);
        float p2 = __expf(a2 - mnew);
        float p3 = __expf(a3 - mnew);
        float ps = (p0 + p1) + (p2 + p3);
        ps += __shfl_xor(ps, 1);
        ps += __shfl_xor(ps, 2);
        ps += __shfl_xor(ps, 4);
        ps += __shfl_xor(ps, 8);
        lrow[mt][r] = lrow[mt][r] * scale + ps;
        oacc[mt][0][r] *= scale; oacc[mt][1][r] *= scale;
        oacc[mt][2][r] *= scale; oacc[mt][3][r] *= scale;
        // stage P (bf16): row = mt*16 + r + 8*hi, key col = kt*16 + ln16
        bf16* prow = pw + (mt * 16 + r + 8 * hi) * 64 + ln16;
        prow[0]  = (bf16)p0;
        prow[16] = (bf16)p1;
        prow[32] = (bf16)p2;
        prow[48] = (bf16)p3;
      }
    }
    asm volatile("s_wait_dscnt 0" ::: "memory");

    // ---- O += P V : two K-steps (32 keys each), V frags shared by m-tiles -
#pragma unroll
    for (int ks = 0; ks < 2; ++ks) {
      v16bf pa0 = load_a(pw + (size_t)ln16 * 64 + ks * 32 + 8 * hi);
      v16bf pa1 = load_a(pw + (size_t)(16 + ln16) * 64 + ks * 32 + 8 * hi);
      const bf16* vrow = vb + (size_t)ln16 * N + key0 + ks * 32 + 8 * hi;
      v16bf vf0 = load_b(vrow);
      v16bf vf1 = load_b(vrow + (size_t)16 * N);
      v16bf vf2 = load_b(vrow + (size_t)32 * N);
      v16bf vf3 = load_b(vrow + (size_t)48 * N);
      oacc[0][0] = wmma_bf16(pa0, vf0, oacc[0][0]);
      oacc[0][1] = wmma_bf16(pa0, vf1, oacc[0][1]);
      oacc[0][2] = wmma_bf16(pa0, vf2, oacc[0][2]);
      oacc[0][3] = wmma_bf16(pa0, vf3, oacc[0][3]);
      oacc[1][0] = wmma_bf16(pa1, vf0, oacc[1][0]);
      oacc[1][1] = wmma_bf16(pa1, vf1, oacc[1][1]);
      oacc[1][2] = wmma_bf16(pa1, vf2, oacc[1][2]);
      oacc[1][3] = wmma_bf16(pa1, vf3, oacc[1][3]);
    }
  }

  // ---- normalize and store o[b, n, h*64 + d] as bf16 ----------------------
#pragma unroll
  for (int mt = 0; mt < 2; ++mt) {
#pragma unroll
    for (int r = 0; r < 8; ++r) {
      int m = mt * 16 + r + 8 * hi;
      float il = 1.0f / lrow[mt][r];
      bf16* orow = o + ((size_t)(br * 2048 + q0 + m)) * 768 + h * 64 + ln16;
      orow[0]  = (bf16)(oacc[mt][0][r] * il);
      orow[16] = (bf16)(oacc[mt][1][r] * il);
      orow[32] = (bf16)(oacc[mt][2][r] * il);
      orow[48] = (bf16)(oacc[mt][3][r] * il);
    }
  }
}

// ---------------------------------------------------------------------------
// Host orchestration
// ---------------------------------------------------------------------------
extern "C" void kernel_launch(void* const* d_in, const int* in_sizes, int n_in,
                              void* d_out, int out_size, void* d_ws, size_t ws_size,
                              hipStream_t stream) {
  const float* x      = (const float*)d_in[0];
  const float* ln1_g  = (const float*)d_in[1];
  const float* ln1_b  = (const float*)d_in[2];
  const float* w_qkv  = (const float*)d_in[3];
  const float* b_qkv  = (const float*)d_in[4];
  const float* w_proj = (const float*)d_in[5];
  const float* b_proj = (const float*)d_in[6];
  const float* ln2_g  = (const float*)d_in[7];
  const float* ln2_b  = (const float*)d_in[8];
  const float* w1     = (const float*)d_in[9];
  const float* b1     = (const float*)d_in[10];
  const float* w2     = (const float*)d_in[11];
  const float* b2     = (const float*)d_in[12];
  float* out = (float*)d_out;

  const int M = 4096, C = 768, HID = 3072, NQKV = 2304, N = 2048;

  char* wp = (char*)d_ws;
  auto alloc = [&](size_t bytes) -> void* {
    void* p = (void*)wp;
    wp += (bytes + 255) & ~(size_t)255;
    return p;
  };
  bf16* wqkvT  = (bf16*)alloc((size_t)NQKV * C * 2);
  bf16* wprojT = (bf16*)alloc((size_t)C * C * 2);
  bf16* w1T    = (bf16*)alloc((size_t)HID * C * 2);
  bf16* w2T    = (bf16*)alloc((size_t)C * HID * 2);
  bf16* h1     = (bf16*)alloc((size_t)M * C * 2);
  bf16* qbuf   = (bf16*)alloc((size_t)M * C * 2);
  bf16* kbuf   = (bf16*)alloc((size_t)M * C * 2);
  bf16* vtbuf  = (bf16*)alloc((size_t)M * C * 2);
  bf16* obuf   = (bf16*)alloc((size_t)M * C * 2);
  float* x1    = (float*)alloc((size_t)M * C * 4);
  bf16* h2     = (bf16*)alloc((size_t)M * C * 2);
  bf16* hid    = (bf16*)alloc((size_t)M * HID * 2);
  (void)ws_size; (void)in_sizes; (void)n_in; (void)out_size;

  auto cvt = [&](const float* in, bf16* o, int K, int Nn) {
    int total = K * Nn;
    conv_t_kernel<<<(total + 255) / 256, 256, 0, stream>>>(in, o, K, Nn);
  };
  cvt(w_qkv, wqkvT, C, NQKV);
  cvt(w_proj, wprojT, C, C);
  cvt(w1, w1T, C, HID);
  cvt(w2, w2T, HID, C);

  // LN1
  ln_kernel<<<M, 256, 0, stream>>>(x, ln1_g, ln1_b, h1);

  // QKV projection + scatter (q scaled by 1/sqrt(64)).  128x128 block tile.
  gemm_wmma<4, 4, 0><<<dim3(NQKV / 128, M / 128), 128, 0, stream>>>(
      h1, wqkvT, b_qkv, M, NQKV, C,
      nullptr, nullptr, nullptr, qbuf, kbuf, vtbuf);

  // Attention: 32 queries per wave, 128 per block, 64-key chunks.
  attn_kernel<<<dim3(24, N / 128), 128, 0, stream>>>(qbuf, kbuf, vtbuf, obuf, N);

  // Output projection + residual -> x1 (f32).  128x64 block tile.
  gemm_wmma<4, 2, 1><<<dim3(C / 64, M / 128), 128, 0, stream>>>(
      obuf, wprojT, b_proj, M, C, C,
      nullptr, x1, x, nullptr, nullptr, nullptr);

  // LN2
  ln_kernel<<<M, 256, 0, stream>>>(x1, ln2_g, ln2_b, h2);

  // MLP up + exact GELU.  128x128 block tile.
  gemm_wmma<4, 4, 2><<<dim3(HID / 128, M / 128), 128, 0, stream>>>(
      h2, w1T, b1, M, HID, C,
      hid, nullptr, nullptr, nullptr, nullptr, nullptr);

  // MLP down + residual -> d_out (f32).  128x64 block tile.
  gemm_wmma<4, 2, 1><<<dim3(C / 64, M / 128), 128, 0, stream>>>(
      hid, w2T, b2, M, C, HID,
      nullptr, out, x1, nullptr, nullptr, nullptr);
}
